// Spiral_block_89472758710753
// MI455X (gfx1250) — matrix operand order repaired and checked by reading, compiled, not run on hardware
//
#include <hip/hip_runtime.h>
#include <hip/hip_bf16.h>

#define BB 8
#define NN 50000
#define SS 9
#define CC 64
#define OUTC 64
#define GG 8
#define KDIM 640            // (S+1)*C
#define HSTRIDE 648         // 640 + 8 bf16 pad -> 324 dwords/row, 324 % 64 = 4 (no bank conflict)
#define MTILE 32            // nodes per wave (2 x 16-row WMMA tiles)

typedef __attribute__((ext_vector_type(16))) __bf16 v16bf;
typedef __attribute__((ext_vector_type(8)))  float  v8f;

union Frag { v16bf v; uint4 u[2]; };

// ---- helpers -------------------------------------------------------------

__device__ __forceinline__ unsigned pack2bf(float lo, float hi) {
  unsigned a = __float_as_uint(lo), b = __float_as_uint(hi);
  a = (a + 0x7FFFu + ((a >> 16) & 1u)) >> 16;     // RNE fp32 -> bf16
  b = (b + 0x7FFFu + ((b >> 16) & 1u)) >> 16;
  return (a & 0xFFFFu) | (b << 16);
}

// monotone map float -> uint so unsigned atomicMax == float max
__device__ __forceinline__ unsigned mapMax(float f) {
  unsigned b = __float_as_uint(f);
  return (b & 0x80000000u) ? ~b : (b | 0x80000000u);
}
__device__ __forceinline__ float unmapMax(unsigned u) {
  unsigned b = (u & 0x80000000u) ? (u & 0x7FFFFFFFu) : ~u;
  return __uint_as_float(b);
}

// ---- kernel 1: init workspace --------------------------------------------

__global__ void init_ws_kernel(unsigned* vmaxU, float* sums, float* sumsq) {
  int t = threadIdx.x;
  for (int i = t; i < BB * CC; i += 256) vmaxU[i] = 0u;   // 0 < map(any finite)
  if (t < BB * GG) { sums[t] = 0.f; sumsq[t] = 0.f; }
}

// ---- kernel 2: W fp32 -> bf16 --------------------------------------------

__global__ void convert_w_kernel(const float* __restrict__ W, unsigned short* __restrict__ Wbf) {
  int i = blockIdx.x * 256 + threadIdx.x;
  if (i < KDIM * OUTC) {
    unsigned a = __float_as_uint(W[i]);
    Wbf[(i & 63) * KDIM + (i >> 6)] = (unsigned short)((a + 0x7FFFu + ((a >> 16) & 1u)) >> 16);
  }
}

// ---- kernel 3: vmax over nodes per (b, c) --------------------------------

#define VROWS 784   // rows per block chunk (49 iters of 16)

__global__ void vmax_kernel(const float* __restrict__ x, unsigned* __restrict__ vmaxU) {
  __shared__ float4 red[256];
  const int b = blockIdx.y;
  const int t = threadIdx.x;
  const int c4 = (t & 15) * 4;
  const int r  = t >> 4;
  float4 m;
  m.x = m.y = m.z = m.w = -3.4e38f;
  int n0 = blockIdx.x * VROWS;
  for (int j = 0; j < VROWS / 16; ++j) {
    int n = n0 + r + j * 16;
    if (n < NN) {
      const float4 v = *(const float4*)(x + ((b * NN + n) << 6) + c4);
      m.x = fmaxf(m.x, v.x); m.y = fmaxf(m.y, v.y);
      m.z = fmaxf(m.z, v.z); m.w = fmaxf(m.w, v.w);
    }
  }
  red[t] = m;
  __syncthreads();
  if (t < 16) {
    float4 acc = red[t];
    #pragma unroll
    for (int k = 1; k < 16; ++k) {
      float4 v = red[t + 16 * k];
      acc.x = fmaxf(acc.x, v.x); acc.y = fmaxf(acc.y, v.y);
      acc.z = fmaxf(acc.z, v.z); acc.w = fmaxf(acc.w, v.w);
    }
    unsigned* dst = vmaxU + b * CC + t * 4;
    atomicMax(dst + 0, mapMax(acc.x));
    atomicMax(dst + 1, mapMax(acc.y));
    atomicMax(dst + 2, mapMax(acc.z));
    atomicMax(dst + 3, mapMax(acc.w));
  }
}

// ---- kernel 4: gather + bf16 WMMA GEMM + bias + stats --------------------
// one wave per block; 32-node x 64-out tile (two 16-row WMMA tiles sharing
// every B fragment -> halves W traffic from L2); K = 640 in 20 steps of 32

__global__ void __launch_bounds__(32)
gemm_kernel(const float* __restrict__ x, const int* __restrict__ idx,
            const unsigned short* __restrict__ Wbf, const float* __restrict__ bias,
            const unsigned* __restrict__ vmaxU,
            float* __restrict__ out, float* __restrict__ sums, float* __restrict__ sumsq) {
  __shared__ unsigned short hTile[MTILE * HSTRIDE];   // 41472 B
  __shared__ float gsum[GG], gss[GG];

  const int lane  = threadIdx.x;        // wave32
  const int half  = lane >> 4;          // 0 / 1
  const int nl    = lane & 15;
  const int cl    = nl * 4;             // channel offset handled by this lane
  const int b     = blockIdx.y;
  const int node0 = blockIdx.x * MTILE;

  if (lane < GG) { gsum[lane] = 0.f; gss[lane] = 0.f; }

  // ---- stage gathered neighbor features: 32*9 = 288 segments of 64 ch ----
  #pragma unroll 4
  for (int i = 0; i < 144; ++i) {
    int seg   = i * 2 + half;                  // 0..287
    int node  = seg / 9;
    int s     = seg - node * 9;
    int gnode = node0 + node; if (gnode >= NN) gnode = NN - 1;   // clamp tail
    int row   = idx[gnode * 9 + s];
    const float4 v = *(const float4*)(x + ((b * NN + row) << 6) + cl);
    unsigned long long p = (unsigned long long)pack2bf(v.x, v.y)
                         | ((unsigned long long)pack2bf(v.z, v.w) << 32);
    *(unsigned long long*)(&hTile[node * HSTRIDE + s * 64 + cl]) = p;
  }
  // ---- stage virtual node (vmax) into K = 576..639 ----
  {
    const unsigned* vs = vmaxU + b * CC + cl;
    unsigned long long p = (unsigned long long)pack2bf(unmapMax(vs[0]), unmapMax(vs[1]))
                         | ((unsigned long long)pack2bf(unmapMax(vs[2]), unmapMax(vs[3])) << 32);
    #pragma unroll
    for (int i = 0; i < MTILE / 2; ++i) {
      int node = i * 2 + half;
      *(unsigned long long*)(&hTile[node * HSTRIDE + 576 + cl]) = p;
    }
  }
  __syncthreads();

  // ---- WMMA main loop: 20 k-steps, 8 WMMAs each ----
  v8f acc[2][4] = {};
  const int koff = half << 3;           // A layout: lanes16-31 hold K+8
  const uint4* hrow0 = (const uint4*)(&hTile[nl * HSTRIDE]);
  const uint4* hrow1 = (const uint4*)(&hTile[(16 + nl) * HSTRIDE]);

  for (int kk = 0; kk < KDIM / 32; ++kk) {
    const int k0 = kk * 32;
    Frag a0, a1;
    a0.u[0] = hrow0[(k0 + koff) >> 3];        // K = k0+koff   .. +8  (VGPR0-3)
    a0.u[1] = hrow0[(k0 + 16 + koff) >> 3];   // K = k0+16+koff.. +8  (VGPR4-7)
    a1.u[0] = hrow1[(k0 + koff) >> 3];
    a1.u[1] = hrow1[(k0 + 16 + koff) >> 3];
    const uint4* wrow = (const uint4*)(Wbf + nl * KDIM + k0 + koff);  // lane = N col, Wbf is [n][k]
    #pragma unroll
    for (int t = 0; t < 4; ++t) {
      Frag w;
      w.u[0] = wrow[t * 1280];                // K = k0+koff   .. +8, col t*16+nl
      w.u[1] = wrow[t * 1280 + 2];            // K = k0+16+koff.. +8, col t*16+nl
      acc[0][t] = __builtin_amdgcn_wmma_f32_16x16x32_bf16(
          false, a0.v, false, w.v, (short)0, acc[0][t], false, false);
      acc[1][t] = __builtin_amdgcn_wmma_f32_16x16x32_bf16(
          false, a1.v, false, w.v, (short)0, acc[1][t], false, false);
    }
  }

  // ---- bias, store y (pre-norm) into d_out, accumulate group stats ----
  const int mBase = half * 8;                 // C/D layout: lanes16-31 -> M+8
  #pragma unroll
  for (int tile = 0; tile < 2; ++tile) {
    if (node0 + tile * 16 >= NN) break;       // wave-uniform; after all WMMAs
    const int rowBase = node0 + tile * 16 + mBase;
    #pragma unroll
    for (int t = 0; t < 4; ++t) {
      const int col  = t * 16 + nl;
      const float bc = bias[col];
      const int g    = col >> 3;
      float s_ = 0.f, ss_ = 0.f;
      #pragma unroll
      for (int r = 0; r < 8; ++r) {
        float y = acc[tile][t][r] + bc;
        out[((b * NN + rowBase + r) << 6) + col] = y;
        s_  += y;
        ss_ += y * y;
      }
      atomicAdd(&gsum[g], s_);                // ds_add_f32
      atomicAdd(&gss[g],  ss_);
    }
  }
  __syncthreads();
  if (lane < GG)             atomicAdd(&sums[b * GG + lane], gsum[lane]);
  else if (lane < 2 * GG)    atomicAdd(&sumsq[b * GG + (lane - GG)], gss[lane - GG]);
}

// ---- kernel 5: GroupNorm + ReLU in place ---------------------------------

__global__ void norm_kernel(float* __restrict__ out,
                            const float* __restrict__ sums, const float* __restrict__ sumsq,
                            const float* __restrict__ gamma, const float* __restrict__ beta) {
  int i4 = blockIdx.x * 256 + threadIdx.x;
  int flat = i4 * 4;
  if (flat >= BB * NN * OUTC) return;
  const int b    = flat / (NN * OUTC);
  const int col0 = flat & 63;
  const int g    = col0 >> 3;
  const float cnt  = (float)(NN * (OUTC / GG));
  const float mean = sums[b * GG + g] / cnt;
  const float var  = sumsq[b * GG + g] / cnt - mean * mean;
  const float rs   = rsqrtf(var + 1e-5f);
  float4 v = *(float4*)(out + flat);
  v.x = fmaxf(0.f, (v.x - mean) * rs * gamma[col0 + 0] + beta[col0 + 0]);
  v.y = fmaxf(0.f, (v.y - mean) * rs * gamma[col0 + 1] + beta[col0 + 1]);
  v.z = fmaxf(0.f, (v.z - mean) * rs * gamma[col0 + 2] + beta[col0 + 2]);
  v.w = fmaxf(0.f, (v.w - mean) * rs * gamma[col0 + 3] + beta[col0 + 3]);
  *(float4*)(out + flat) = v;
}

// ---- launch ---------------------------------------------------------------

extern "C" void kernel_launch(void* const* d_in, const int* in_sizes, int n_in,
                              void* d_out, int out_size, void* d_ws, size_t ws_size,
                              hipStream_t stream) {
  const float* x     = (const float*)d_in[0];
  const int*   idx   = (const int*)d_in[1];
  const float* W     = (const float*)d_in[2];
  const float* bias  = (const float*)d_in[3];
  const float* gamma = (const float*)d_in[4];
  const float* beta  = (const float*)d_in[5];
  float* out = (float*)d_out;

  unsigned char* ws = (unsigned char*)d_ws;
  unsigned*       vmaxU = (unsigned*)ws;                       // 512 u32
  float*          sums  = (float*)(ws + 2048);                 // 64 f32
  float*          sumsq = (float*)(ws + 2048 + 256);           // 64 f32
  unsigned short* Wbf   = (unsigned short*)(ws + 4096);        // 40960 bf16

  init_ws_kernel<<<1, 256, 0, stream>>>(vmaxU, sums, sumsq);
  convert_w_kernel<<<(KDIM * OUTC + 255) / 256, 256, 0, stream>>>(W, Wbf);
  vmax_kernel<<<dim3((NN + VROWS - 1) / VROWS, BB), 256, 0, stream>>>(x, vmaxU);
  gemm_kernel<<<dim3((NN + MTILE - 1) / MTILE, BB), 32, 0, stream>>>(x, idx, Wbf, bias, vmaxU,
                                                                     out, sums, sumsq);
  norm_kernel<<<(BB * NN * OUTC / 4 + 255) / 256, 256, 0, stream>>>(out, sums, sumsq, gamma, beta);
}
